// ImageTriplaneGenerator_67276367724843
// MI455X (gfx1250) — compile-verified, matrix operand order checked by reading, not run on hardware
//
#include <hip/hip_runtime.h>
#include <hip/hip_bf16.h>
#include <math.h>

// ---------------------------------------------------------------------------
// Problem constants (from the reference)
// ---------------------------------------------------------------------------
constexpr int kR  = 128;
constexpr int kC  = 64;
constexpr int kS  = 3;
constexpr int kNL = 2;
constexpr int kNH = 4;
constexpr int kB  = 2;
constexpr int kV  = 4;
constexpr int kN  = kR * kR;        // 16384
constexpr int kHD = kC / kNH;       // 16
constexpr int kT  = 2 * kS;         // 6
constexpr int kM  = kB * kN;        // 32768 query rows
constexpr int kMT = kM * kT;        // 196608 context rows

typedef __attribute__((ext_vector_type(16))) _Float16 v16h;
typedef __attribute__((ext_vector_type(8)))  float    v8f;
typedef unsigned int u32x4 __attribute__((ext_vector_type(4)));
typedef int          i32x8 __attribute__((ext_vector_type(8)));
typedef int          i32x4 __attribute__((ext_vector_type(4)));

union V16H { v16h v; uint4 u[2]; };

#define WMMA_F16(a, b, c) \
  __builtin_amdgcn_wmma_f32_16x16x32_f16(false, (a), false, (b), (short)0, (c), false, false)

// ---------------------------------------------------------------------------
// Utility kernels
// ---------------------------------------------------------------------------
__global__ void k_zero(float* __restrict__ p, size_t n) {
  size_t i = (size_t)blockIdx.x * blockDim.x + threadIdx.x;
  size_t stride = (size_t)gridDim.x * blockDim.x;
  for (; i < n; i += stride) p[i] = 0.f;
}

__global__ void k_f32_to_f16(const float* __restrict__ s, _Float16* __restrict__ d, size_t n) {
  size_t i = (size_t)blockIdx.x * blockDim.x + threadIdx.x;
  size_t stride = (size_t)gridDim.x * blockDim.x;
  for (; i < n; i += stride) d[i] = (_Float16)s[i];
}

// ---------------------------------------------------------------------------
// Splat: scatter-add with trilinear weights (one v-slice at a time so the
// accumulators stay at 25 MB and live in L2). block = one (b,n) point,
// thread = channel. Plane/corner lists match the reference's plane_accum.
// ---------------------------------------------------------------------------
__global__ void k_splat_scatter(const float* __restrict__ img,   // (B,V,C,N)
                                const float* __restrict__ pts,   // (B,V,N,3)
                                int vIdx,
                                float* __restrict__ accF,        // (3,B,N,C)
                                float* __restrict__ accW)        // (3,B,N)
{
  int blk = blockIdx.x;               // b*N + n
  int b = blk / kN, n = blk % kN;
  int ch = threadIdx.x;               // 0..63
  const float* p = pts + (((size_t)(b * kV + vIdx)) * kN + n) * 3;
  const float hi = (float)(kR - 1) - 1e-5f;
  float gx = fminf(fmaxf((p[0] + 1.f) * (kR - 1) * 0.5f, 0.f), hi);
  float gy = fminf(fmaxf((p[1] + 1.f) * (kR - 1) * 0.5f, 0.f), hi);
  float gz = fminf(fmaxf((p[2] + 1.f) * (kR - 1) * 0.5f, 0.f), hi);
  int x0 = (int)floorf(gx), y0 = (int)floorf(gy), z0 = (int)floorf(gz);
  float wx = gx - x0, wy = gy - y0, wz = gz - z0;
  int x1 = min(x0 + 1, kR - 1), y1 = min(y0 + 1, kR - 1), z1 = min(z0 + 1, kR - 1);
  float w000 = (1 - wx) * (1 - wy) * (1 - wz);
  float w001 = (1 - wx) * (1 - wy) * wz;
  float w010 = (1 - wx) * wy * (1 - wz);
  float w100 = wx * (1 - wy) * (1 - wz);
  float w110 = wx * wy * (1 - wz);
  float w111 = wx * wy * wz;

  const int   pa[12] = {0,0,0,0, 1,1,1,1, 2,2,2,2};
  const int   ia[12] = {x0,x0,x1,x1, x0,x0,x1,x1, y0,y0,y1,y1};
  const int   ib[12] = {y0,y1,y0,y1, z0,z1,z0,z1, z0,z1,z0,z1};
  const float ww[12] = {w000,w010,w100,w110, w000,w001,w100,w111, w000,w001,w010,w111};

  float fv = img[(((size_t)(b * kV + vIdx)) * kC + ch) * kN + n];
  #pragma unroll
  for (int e = 0; e < 12; ++e) {
    size_t cell = ((size_t)(pa[e] * kB + b)) * kN + (size_t)ia[e] * kR + ib[e];
    atomicAdd(&accF[cell * kC + ch], ww[e] * fv);
    if (ch == 0) atomicAdd(&accW[cell], ww[e]);
  }
}

__global__ void k_splat_norm_add(const float* __restrict__ accF,
                                 const float* __restrict__ accW,
                                 float* __restrict__ plane)     // (3,B,N,C)
{
  size_t total = (size_t)3 * kB * kN * kC;
  size_t i = (size_t)blockIdx.x * blockDim.x + threadIdx.x;
  size_t stride = (size_t)gridDim.x * blockDim.x;
  for (; i < total; i += stride) {
    size_t cell = i / kC;
    plane[i] += accF[i] / fmaxf(accW[cell], 1e-8f) * (1.f / kV);
  }
}

__global__ void k_add_query(float* __restrict__ plane, const float* __restrict__ q) {
  size_t total = (size_t)3 * kB * kN * kC;
  size_t i = (size_t)blockIdx.x * blockDim.x + threadIdx.x;
  size_t stride = (size_t)gridDim.x * blockDim.x;
  for (; i < total; i += stride) {
    int ch = (int)(i % kC);
    size_t j = i / kC;
    int n = (int)(j % kN);
    size_t m2 = j / kN;
    int pl = (int)(m2 / kB);
    plane[i] += q[((size_t)pl * kN + n) * kC + ch];
  }
}

// ---------------------------------------------------------------------------
// Context gather + LayerNorm fused: one wave32 per (b,n,t) token; 2 ch/lane,
// wave shuffle reduction for mean/var; writes f16 rows ready for WMMA.
// ---------------------------------------------------------------------------
__global__ void k_ctx_ln(const float* __restrict__ planes,   // (3,B,N,C)
                         int curPlane,
                         const float* __restrict__ g, const float* __restrict__ bta,
                         _Float16* __restrict__ cn)           // (B,N,T,C) f16
{
  int wid = blockIdx.x * (blockDim.x >> 5) + (threadIdx.x >> 5);
  int lane = threadIdx.x & 31;
  if (wid >= kB * kN * kT) return;
  int t = wid % kT;
  int n = (wid / kT) % kN;
  int b = wid / (kT * kN);
  int r = n / kR, c = n % kR;
  float linc = -1.f + 2.f * c / (kR - 1);
  float linr = -1.f + 2.f * r / (kR - 1);
  int kk = (t < kS) ? t : (t - kS);
  float sv = -1.f + (float)kk;            // S=3 -> {-1,0,1}
  int src; float c0, c1;
  if (curPlane == 0)      { if (t < kS) { src = 1; c0 = linc; c1 = sv; } else { src = 2; c0 = linr; c1 = sv;   } }
  else if (curPlane == 1) { if (t < kS) { src = 0; c0 = linc; c1 = sv; } else { src = 2; c0 = sv;   c1 = linr; } }
  else                    { if (t < kS) { src = 0; c0 = sv;   c1 = linc;} else { src = 1; c0 = sv;   c1 = linr; } }

  const float Rm1 = (float)(kR - 1);
  float gc0 = fminf(fmaxf((c0 * 0.5f + 0.5f) * Rm1, 0.f), Rm1);
  float gc1 = fminf(fmaxf((c1 * 0.5f + 0.5f) * Rm1, 0.f), Rm1);
  float u = fminf(fmaxf((gc0 + 1.f) * 0.5f * Rm1, 0.f), Rm1);   // width (col)
  float w = fminf(fmaxf((gc1 + 1.f) * 0.5f * Rm1, 0.f), Rm1);   // height (row)
  int x0 = (int)floorf(u), y0 = (int)floorf(w);
  int x1 = min(x0 + 1, kR - 1), y1 = min(y0 + 1, kR - 1);
  float wx = u - x0, wyf = w - y0;

  const float* base = planes + ((size_t)src * kB + b) * (size_t)kN * kC;
  const float* r00 = base + ((size_t)y0 * kR + x0) * kC;
  const float* r01 = base + ((size_t)y0 * kR + x1) * kC;
  const float* r10 = base + ((size_t)y1 * kR + x0) * kC;
  const float* r11 = base + ((size_t)y1 * kR + x1) * kC;

  int ch0 = lane, ch1 = lane + 32;
  float a0 = (1 - wyf) * ((1 - wx) * r00[ch0] + wx * r01[ch0]) + wyf * ((1 - wx) * r10[ch0] + wx * r11[ch0]);
  float a1 = (1 - wyf) * ((1 - wx) * r00[ch1] + wx * r01[ch1]) + wyf * ((1 - wx) * r10[ch1] + wx * r11[ch1]);

  float s1 = a0 + a1, s2 = a0 * a0 + a1 * a1;
  #pragma unroll
  for (int o = 16; o >= 1; o >>= 1) { s1 += __shfl_xor(s1, o, 32); s2 += __shfl_xor(s2, o, 32); }
  float mean = s1 * (1.f / kC);
  float var  = s2 * (1.f / kC) - mean * mean;
  float rstd = rsqrtf(var + 1e-5f);
  size_t row = (size_t)wid * kC;
  cn[row + ch0] = (_Float16)((a0 - mean) * rstd * g[ch0] + bta[ch0]);
  cn[row + ch1] = (_Float16)((a1 - mean) * rstd * g[ch1] + bta[ch1]);
}

// LayerNorm over fp32 rows [Mrows, 64] -> f16, wave per row
__global__ void k_ln_rows(const float* __restrict__ x,
                          const float* __restrict__ g, const float* __restrict__ bta,
                          _Float16* __restrict__ out, int Mrows)
{
  int wid = blockIdx.x * (blockDim.x >> 5) + (threadIdx.x >> 5);
  int lane = threadIdx.x & 31;
  if (wid >= Mrows) return;
  const float* rowp = x + (size_t)wid * kC;
  int ch0 = lane, ch1 = lane + 32;
  float a0 = rowp[ch0], a1 = rowp[ch1];
  float s1 = a0 + a1, s2 = a0 * a0 + a1 * a1;
  #pragma unroll
  for (int o = 16; o >= 1; o >>= 1) { s1 += __shfl_xor(s1, o, 32); s2 += __shfl_xor(s2, o, 32); }
  float mean = s1 * (1.f / kC);
  float var  = s2 * (1.f / kC) - mean * mean;
  float rstd = rsqrtf(var + 1e-5f);
  _Float16* orow = out + (size_t)wid * kC;
  orow[ch0] = (_Float16)((a0 - mean) * rstd * g[ch0] + bta[ch0]);
  orow[ch1] = (_Float16)((a1 - mean) * rstd * g[ch1] + bta[ch1]);
}

// ---------------------------------------------------------------------------
// TDM helper: DMA a contiguous run of `totalHalves` f16 values from global
// memory into LDS offset 0 (the block's weight slice). 1-D D# per
// CDNA5 ISA: data_size=2B, tile_dim0=totalHalves,
// tensor_dim0(_stride)=totalHalves, type=2. Issued by one lane (TDM ignores
// EXEC), completion via s_wait_tensorcnt, published by the block barrier.
// ---------------------------------------------------------------------------
#if __has_builtin(__builtin_amdgcn_tensor_load_to_lds)
#define USE_TDM 1
__device__ __forceinline__ void tdm_load_lds0(const _Float16* gsrc, unsigned totalHalves) {
  unsigned long long ga = (unsigned long long)(size_t)gsrc;
  u32x4 g0;
  g0[0] = 1u;                                            // count=1, user D#
  g0[1] = 0u;                                            // lds_addr = 0
  g0[2] = (unsigned)(ga & 0xFFFFFFFFu);                  // global_addr lo
  g0[3] = (unsigned)((ga >> 32) & 0x1FFFFFFu) | (2u << 30); // addr hi | type=2
  i32x8 g1;
  g1[0] = 0x10000;                                       // data_size=2B, no mask
  g1[1] = (int)((totalHalves & 0xFFFFu) << 16);          // tensor_dim0 [63:48]
  g1[2] = (int)(((totalHalves >> 16) & 0xFFFFu) | (1u << 16)); // dim0 hi | tensor_dim1=1
  g1[3] = (int)(totalHalves << 16);                      // tile_dim0 [127:112]
  g1[4] = 1;                                             // tile_dim1 = 1
  g1[5] = (int)totalHalves;                              // tensor_dim0_stride
  g1[6] = 0;
  g1[7] = 0;
  i32x4 z4 = {0, 0, 0, 0};
#if defined(__clang_major__) && __clang_major__ >= 23
  i32x8 z8 = {0, 0, 0, 0, 0, 0, 0, 0};
  __builtin_amdgcn_tensor_load_to_lds(g0, g1, z4, z4, z8, 0);
#else
  __builtin_amdgcn_tensor_load_to_lds(g0, g1, z4, z4, 0);
#endif
  __builtin_amdgcn_s_wait_tensorcnt(0);
}
#else
#define USE_TDM 0
#endif

// ---------------------------------------------------------------------------
// WMMA GEMM: D = A[M,K] @ W[N,K]^T + bias, f16 in / f32 accumulate.
// Block = 256 thr = 8 waves covering 128 rows x 64 cols. Each wave holds FOUR
// 16x16 accumulators; per K-step: the four B tiles are loaded into DISTINCT
// registers (8 ds_load_b128 in one clause, one wait) so the 4 v_wmma issue
// back-to-back, and the next A tile is software-pipelined one K-step ahead
// so its global-load latency hides under the MMA group. Weight slice (<=32KB)
// is DMA'd into LDS by the Tensor Data Mover (fallback: cooperative copy).
// modes: 0 = store f16; 1 = exact-GELU then store f16; 2 = fp32 residual add.
// ---------------------------------------------------------------------------
constexpr int kNCHUNK = 64;          // output cols per block

__global__ void __launch_bounds__(256)
k_gemm_wmma(const _Float16* __restrict__ A,
            const _Float16* __restrict__ W,
            const float* __restrict__ bias,
            int Mrows, int Ncols, int K, int mode,
            _Float16* __restrict__ outH,
            const float* __restrict__ residIn,
            float* __restrict__ outF)
{
  __shared__ alignas(16) _Float16 Ws[kNCHUNK * 256];   // 32 KB max (K<=256)
  int nbase = blockIdx.y * kNCHUNK;
  unsigned sliceHalves = (unsigned)(kNCHUNK * K);
#if USE_TDM
  if (threadIdx.x == 0) {
    tdm_load_lds0(W + (size_t)nbase * K, sliceHalves);
  }
  asm volatile("" : : "r"(&Ws[0]) : "memory");   // LDS written by TDM, keep live
#else
  { // cooperative weight-slice copy (contiguous kNCHUNK*K halves)
    const uint4* wsrc = (const uint4*)(W + (size_t)nbase * K);
    int nvec = (int)(sliceHalves >> 3);
    for (int i = threadIdx.x; i < nvec; i += 256) ((uint4*)Ws)[i] = wsrc[i];
  }
#endif
  __syncthreads();

  int wave = threadIdx.x >> 5, lane = threadIdx.x & 31;
  int mbase = blockIdx.x * 128 + wave * 16;
  int lm = lane & 15;
  int khalf = (lane >> 4) << 3;                       // 0 or 8
  const _Float16* Arow = A + (size_t)(mbase + lm) * K;
  const _Float16* Wr0 = Ws + (size_t)(lm +  0) * K;
  const _Float16* Wr1 = Ws + (size_t)(lm + 16) * K;
  const _Float16* Wr2 = Ws + (size_t)(lm + 32) * K;
  const _Float16* Wr3 = Ws + (size_t)(lm + 48) * K;

  v8f acc0 = {}, acc1 = {}, acc2 = {}, acc3 = {};

  // prologue: first A tile in flight
  V16H a;
  a.u[0] = *(const uint4*)(Arow + khalf);
  a.u[1] = *(const uint4*)(Arow + khalf + 16);
  __builtin_prefetch(Arow + khalf + 32, 0, 1);         // global_prefetch_b8

  for (int kt = 0; kt < K; kt += 32) {
    int kb = kt + khalf;
    // software-pipelined next A tile (hidden under this step's 4 WMMAs)
    V16H an;
    if (kt + 32 < K) {
      an.u[0] = *(const uint4*)(Arow + kb + 32);
      an.u[1] = *(const uint4*)(Arow + kb + 48);
      __builtin_prefetch(Arow + kb + 64, 0, 1);
    } else {
      an = a;
    }
    // four distinct B tiles -> one ds clause, one wait, 4 back-to-back WMMAs
    V16H b0, b1, b2, b3;
    b0.u[0] = *(const uint4*)(Wr0 + kb); b0.u[1] = *(const uint4*)(Wr0 + kb + 16);
    b1.u[0] = *(const uint4*)(Wr1 + kb); b1.u[1] = *(const uint4*)(Wr1 + kb + 16);
    b2.u[0] = *(const uint4*)(Wr2 + kb); b2.u[1] = *(const uint4*)(Wr2 + kb + 16);
    b3.u[0] = *(const uint4*)(Wr3 + kb); b3.u[1] = *(const uint4*)(Wr3 + kb + 16);
    acc0 = WMMA_F16(a.v, b0.v, acc0);
    acc1 = WMMA_F16(a.v, b1.v, acc1);
    acc2 = WMMA_F16(a.v, b2.v, acc2);
    acc3 = WMMA_F16(a.v, b3.v, acc3);
    a = an;
  }

  v8f accA[4] = {acc0, acc1, acc2, acc3};
  int mo = (lane >> 4) << 3;                           // 0 or 8 (D layout)
  #pragma unroll
  for (int j = 0; j < 4; ++j) {
    int ncol = nbase + j * 16 + lm;
    float bn = bias[ncol];
    #pragma unroll
    for (int vv = 0; vv < 8; ++vv) {
      int m = mbase + mo + vv;
      if (m >= Mrows) continue;
      float val = accA[j][vv] + bn;
      if (mode == 0) {
        outH[(size_t)m * Ncols + ncol] = (_Float16)val;
      } else if (mode == 1) {
        val = 0.5f * val * (1.f + erff(val * 0.70710678118f));
        outH[(size_t)m * Ncols + ncol] = (_Float16)val;
      } else {
        outF[(size_t)m * Ncols + ncol] = residIn[(size_t)m * Ncols + ncol] + val;
      }
    }
  }
}

// ---------------------------------------------------------------------------
// Tiny-context attention: thread per (query m, head h); T=6, HD=16.
// ---------------------------------------------------------------------------
__global__ void k_attention(const _Float16* __restrict__ q,
                            const _Float16* __restrict__ kmat,
                            const _Float16* __restrict__ vmat,
                            _Float16* __restrict__ att)
{
  int tid = blockIdx.x * blockDim.x + threadIdx.x;
  if (tid >= kM * kNH) return;
  int h = tid & (kNH - 1);
  int m = tid >> 2;
  const _Float16* qp = q + (size_t)m * kC + h * kHD;
  float qf[kHD];
  #pragma unroll
  for (int d = 0; d < kHD; ++d) qf[d] = (float)qp[d];
  float sc[kT]; float mx = -1e30f;
  #pragma unroll
  for (int t = 0; t < kT; ++t) {
    const _Float16* kp = kmat + ((size_t)m * kT + t) * kC + h * kHD;
    float s = 0.f;
    #pragma unroll
    for (int d = 0; d < kHD; ++d) s += qf[d] * (float)kp[d];
    s *= 0.25f;                              // 1/sqrt(16)
    sc[t] = s; mx = fmaxf(mx, s);
  }
  float den = 0.f;
  #pragma unroll
  for (int t = 0; t < kT; ++t) { sc[t] = __expf(sc[t] - mx); den += sc[t]; }
  float inv = 1.f / den;
  float o[kHD];
  #pragma unroll
  for (int d = 0; d < kHD; ++d) o[d] = 0.f;
  #pragma unroll
  for (int t = 0; t < kT; ++t) {
    float pw = sc[t] * inv;
    const _Float16* vp = vmat + ((size_t)m * kT + t) * kC + h * kHD;
    #pragma unroll
    for (int d = 0; d < kHD; ++d) o[d] += pw * (float)vp[d];
  }
  _Float16* op = att + (size_t)m * kC + h * kHD;
  #pragma unroll
  for (int d = 0; d < kHD; ++d) op[d] = (_Float16)o[d];
}

// Final transpose: (3,B,N,C) plane layout -> (B,3,C,R,R)
__global__ void k_writeout(const float* __restrict__ plane, float* __restrict__ out) {
  size_t total = (size_t)3 * kB * kN * kC;
  size_t i = (size_t)blockIdx.x * blockDim.x + threadIdx.x;
  size_t stride = (size_t)gridDim.x * blockDim.x;
  for (; i < total; i += stride) {
    int n = (int)(i % kN);
    size_t u = i / kN;
    int ch = (int)(u % kC);
    size_t u2 = u / kC;
    int pl = (int)(u2 % 3);
    int b  = (int)(u2 / 3);
    out[i] = plane[(((size_t)pl * kB + b) * kN + n) * kC + ch];
  }
}

// ---------------------------------------------------------------------------
// Host orchestration
// ---------------------------------------------------------------------------
extern "C" void kernel_launch(void* const* d_in, const int* in_sizes, int n_in,
                              void* d_out, int out_size, void* d_ws, size_t ws_size,
                              hipStream_t stream) {
  (void)in_sizes; (void)n_in; (void)out_size; (void)ws_size;
  const float* img   = (const float*)d_in[0];
  const float* pts   = (const float*)d_in[1];
  const float* query = (const float*)d_in[2];
  const float* ln1g  = (const float*)d_in[3];
  const float* ln1b  = (const float*)d_in[4];
  const float* ln2g  = (const float*)d_in[5];
  const float* ln2b  = (const float*)d_in[6];
  const float* inw   = (const float*)d_in[7];
  const float* inb   = (const float*)d_in[8];
  const float* outw  = (const float*)d_in[9];
  const float* outb  = (const float*)d_in[10];
  const float* fw1   = (const float*)d_in[11];
  const float* fb1   = (const float*)d_in[12];
  const float* fw2   = (const float*)d_in[13];
  const float* fb2   = (const float*)d_in[14];

  char* wp = (char*)d_ws;
  auto alloc = [&](size_t bytes) -> void* {
    void* r = (void*)wp; wp += (bytes + 255) & ~(size_t)255; return r;
  };
  const size_t planeElems = (size_t)3 * kB * kN * kC;          // 6.29M
  float*    planeA = (float*)alloc(planeElems * 4);            // 25.2 MB
  float*    planeB = (float*)alloc(planeElems * 4);            // 25.2 MB
  float*    accF   = (float*)alloc(planeElems * 4);            // 25.2 MB
  float*    accW   = (float*)alloc((size_t)3 * kB * kN * 4);   // 0.39 MB
  _Float16* cn     = (_Float16*)alloc((size_t)kMT * kC * 2);   // 25.2 MB
  _Float16* xn     = (_Float16*)alloc((size_t)kM * kC * 2);    // 4.2 MB
  _Float16* qb     = (_Float16*)alloc((size_t)kM * kC * 2);
  _Float16* kb     = (_Float16*)alloc((size_t)kMT * kC * 2);   // 25.2 MB
  _Float16* vb     = (_Float16*)alloc((size_t)kMT * kC * 2);   // 25.2 MB
  _Float16* attb   = (_Float16*)alloc((size_t)kM * kC * 2);
  _Float16* h1     = (_Float16*)alloc((size_t)kM * 4 * kC * 2);// 16.8 MB
  float*    xwork  = (float*)alloc((size_t)kM * kC * 4);       // 8.4 MB
  _Float16* inw16  = (_Float16*)alloc((size_t)kNL * 3 * kC * kC * 2);
  _Float16* outw16 = (_Float16*)alloc((size_t)kNL * kC * kC * 2);
  _Float16* w116   = (_Float16*)alloc((size_t)kNL * 4 * kC * kC * 2);
  _Float16* w216   = (_Float16*)alloc((size_t)kNL * 4 * kC * kC * 2);

  // --- Weight conversion (fp32 -> f16) ---
  auto cvt = [&](const float* s, _Float16* d, size_t n) {
    k_f32_to_f16<<<dim3((unsigned)((n + 255) / 256)), dim3(256), 0, stream>>>(s, d, n);
  };
  cvt(inw,  inw16,  (size_t)kNL * 3 * kC * kC);
  cvt(outw, outw16, (size_t)kNL * kC * kC);
  cvt(fw1,  w116,   (size_t)kNL * 4 * kC * kC);
  cvt(fw2,  w216,   (size_t)kNL * 4 * kC * kC);

  // --- Splat phase ---
  k_zero<<<dim3(4096), dim3(256), 0, stream>>>(planeA, planeElems);
  for (int v = 0; v < kV; ++v) {
    k_zero<<<dim3(4096), dim3(256), 0, stream>>>(accF, planeElems);
    k_zero<<<dim3(512),  dim3(256), 0, stream>>>(accW, (size_t)3 * kB * kN);
    k_splat_scatter<<<dim3(kB * kN), dim3(kC), 0, stream>>>(img, pts, v, accF, accW);
    k_splat_norm_add<<<dim3(4096), dim3(256), 0, stream>>>(accF, accW, planeA);
  }
  k_add_query<<<dim3(4096), dim3(256), 0, stream>>>(planeA, query);

  // --- Transformer layers (ping-pong planes: ctx reads prev-iteration feats) ---
  float* cur = planeA;
  float* nxt = planeB;
  for (int l = 0; l < kNL; ++l) {
    const _Float16* inw16l = inw16 + (size_t)l * 3 * kC * kC;
    for (int pl = 0; pl < 3; ++pl) {
      float* x = cur + (size_t)pl * kB * kN * kC;
      // ctx gather + LN1 -> cn (f16)
      k_ctx_ln<<<dim3((kB * kN * kT) / 8), dim3(256), 0, stream>>>(
          cur, pl, ln1g + l * kC, ln1b + l * kC, cn);
      // LN1(x) -> xn (f16)
      k_ln_rows<<<dim3(kM / 8), dim3(256), 0, stream>>>(x, ln1g + l * kC, ln1b + l * kC, xn, kM);
      // q = xn @ wq^T + bq
      dim3 gq(kM / 128, kC / kNCHUNK);                  // (256, 1)
      k_gemm_wmma<<<gq, 256, 0, stream>>>(xn, inw16l, inb + (size_t)l * 3 * kC,
                                          kM, kC, kC, 0, qb, nullptr, nullptr);
      // k,v = cn @ wk^T/wv^T
      dim3 gkv(kMT / 128, kC / kNCHUNK);                // (1536, 1)
      k_gemm_wmma<<<gkv, 256, 0, stream>>>(cn, inw16l + (size_t)kC * kC,
                                           inb + (size_t)l * 3 * kC + kC,
                                           kMT, kC, kC, 0, kb, nullptr, nullptr);
      k_gemm_wmma<<<gkv, 256, 0, stream>>>(cn, inw16l + (size_t)2 * kC * kC,
                                           inb + (size_t)l * 3 * kC + 2 * kC,
                                           kMT, kC, kC, 0, vb, nullptr, nullptr);
      // attention -> attb (f16)
      k_attention<<<dim3((kM * kNH) / 256), dim3(256), 0, stream>>>(qb, kb, vb, attb);
      // out-proj + residual -> xwork (fp32)
      k_gemm_wmma<<<gq, 256, 0, stream>>>(attb, outw16 + (size_t)l * kC * kC,
                                          outb + (size_t)l * kC,
                                          kM, kC, kC, 2, nullptr, x, xwork);
      // LN2 -> xn (reused, f16)
      k_ln_rows<<<dim3(kM / 8), dim3(256), 0, stream>>>(xwork, ln2g + l * kC, ln2b + l * kC, xn, kM);
      // FFN1 (+GELU) -> h1 (f16)
      dim3 gf1(kM / 128, (4 * kC) / kNCHUNK);           // (256, 4)
      k_gemm_wmma<<<gf1, 256, 0, stream>>>(xn, w116 + (size_t)l * 4 * kC * kC,
                                           fb1 + (size_t)l * 4 * kC,
                                           kM, 4 * kC, kC, 1, h1, nullptr, nullptr);
      // FFN2 + residual -> nxt plane (fp32)
      k_gemm_wmma<<<gq, 256, 0, stream>>>(h1, w216 + (size_t)l * 4 * kC * kC,
                                          fb2 + (size_t)l * kC,
                                          kM, kC, 4 * kC, 2, nullptr, xwork,
                                          nxt + (size_t)pl * kB * kN * kC);
    }
    float* tmp = cur; cur = nxt; nxt = tmp;
  }

  // --- Final layout transform ---
  k_writeout<<<dim3(4096), dim3(256), 0, stream>>>(cur, (float*)d_out);
}
